// BahdanauAttention_21079699488857
// MI455X (gfx1250) — matrix operand order repaired and checked by reading, not compiled, not run on hardware
//
#include <hip/hip_runtime.h>
#include <stddef.h>

// Problem dims (fixed by the reference)
#define BB 32
#define SS 2048
#define DD 1024
#define UU 1024

typedef __attribute__((ext_vector_type(16))) __bf16          v16bf;
typedef __attribute__((ext_vector_type(8)))  float           v8f;
typedef __attribute__((ext_vector_type(8)))  unsigned short  u16x8;

struct FragBits { u16x8 lo; u16x8 hi; };

static __device__ __forceinline__ v16bf mkfrag(u16x8 lo, u16x8 hi) {
    FragBits f{lo, hi};
    return __builtin_bit_cast(v16bf, f);
}

// float -> bf16 round-to-nearest-even (bit twiddle; NaN edge ignored for speed)
static __device__ __forceinline__ unsigned short f2bf(float f) {
    unsigned int u = __builtin_bit_cast(unsigned int, f);
    unsigned int r = u + 0x7FFFu + ((u >> 16) & 1u);
    return (unsigned short)(r >> 16);
}

// Native CDNA5 V_TANH_F32 (TRANS op, co-executes with WMMA pipe) with fallback
static __device__ __forceinline__ float tanh_fast(float x) {
#if __has_builtin(__builtin_amdgcn_tanh_f32)
    return __builtin_amdgcn_tanh_f32(x);
#elif __has_builtin(__builtin_amdgcn_tanhf)
    return __builtin_amdgcn_tanhf(x);
#else
    return tanhf(x);
#endif
}

// ---------------------------------------------------------------------------
// Kernel 1: W1 [D,U] f32  ->  W1t [U,D] bf16 (LDS-tiled transpose)
// ---------------------------------------------------------------------------
__global__ __launch_bounds__(256) void k_w1t(const float* __restrict__ W1,
                                             unsigned short* __restrict__ W1t) {
    __shared__ float tile[32][33];
    const int bu = blockIdx.x * 32;   // u tile
    const int bd = blockIdx.y * 32;   // d tile
    const int tx = threadIdx.x;       // 0..31
    const int ty = threadIdx.y;       // 0..7
    #pragma unroll
    for (int i = 0; i < 32; i += 8)
        tile[ty + i][tx] = W1[(size_t)(bd + ty + i) * UU + (bu + tx)];
    __syncthreads();
    #pragma unroll
    for (int i = 0; i < 32; i += 8)
        W1t[(size_t)(bu + ty + i) * DD + (bd + tx)] = f2bf(tile[tx][ty + i]);
}

// ---------------------------------------------------------------------------
// Kernel 2: proj_q[b,u] = query[b,:] @ W2[:,u] + b1[u] + b2[u]   (biases folded)
// ---------------------------------------------------------------------------
__global__ __launch_bounds__(256) void k_projq(const float* __restrict__ query,
                                               const float* __restrict__ W2,
                                               const float* __restrict__ b1,
                                               const float* __restrict__ b2,
                                               float* __restrict__ pq) {
    const int u = blockIdx.x * 256 + threadIdx.x;
    const int b = blockIdx.y;
    const float* qr = query + (size_t)b * DD;
    float acc = b1[u] + b2[u];
    #pragma unroll 4
    for (int d = 0; d < DD; ++d)
        acc = fmaf(qr[d], W2[(size_t)d * UU + u], acc);
    pq[(size_t)b * UU + u] = acc;
}

// ---------------------------------------------------------------------------
// Kernel 3 (the WMMA core):
//   scores[b,s] = V . tanh(values[b,s,:] @ W1 + proj_q[b,:]) + bV
// WG = 256 thr (8 waves), owns 64 flat rows of [B*S]. values rows staged once
// to LDS as bf16. Wave w: s-tile = w&3, u-half = w>>2 (32 u-tiles each).
// ---------------------------------------------------------------------------
#define ROWP 1032   // padded LDS row stride in halves (1024 + 8) -> bank spread

__global__ __launch_bounds__(256) void k_scores(const float* __restrict__ values,
                                                const unsigned short* __restrict__ W1t,
                                                const float* __restrict__ pq,
                                                const float* __restrict__ Vvec,
                                                const float* __restrict__ bV,
                                                float* __restrict__ scores) {
    __shared__ unsigned short sA[64 * ROWP];   // ~132 KB bf16 A-tile
    __shared__ float sScore[64];

    const int tid  = threadIdx.x;
    const int row0 = blockIdx.x * 64;          // flat row base in [B*S]
    const int b    = row0 >> 11;               // row0 / S (64 | 2048, no straddle)

    // ---- stage 64x1024 values rows: f32 global -> bf16 LDS ----
    const float4* vals4 = (const float4*)(values + (size_t)row0 * DD);
    for (int i = tid; i < 64 * 256; i += 256) {
        const int r  = i >> 8;       // row 0..63
        const int c4 = i & 255;      // float4 column
        float4 v = vals4[(size_t)r * 256 + c4];
        const int o = r * ROWP + c4 * 4;
        sA[o + 0] = f2bf(v.x);
        sA[o + 1] = f2bf(v.y);
        sA[o + 2] = f2bf(v.z);
        sA[o + 3] = f2bf(v.w);
    }
    if (tid < 64) sScore[tid] = 0.0f;
    __syncthreads();

    const int lane  = tid & 31;
    const int wave  = tid >> 5;
    const int m     = lane & 15;     // row within tile (A) / column (B)
    const int sel   = lane >> 4;     // lane-half: K-window select
    const int stile = wave & 3;      // which 16-row s-tile
    const int usel  = wave >> 2;     // which half of the 64 u-tiles

    const unsigned rbase = (unsigned)((stile * 16 + m) * ROWP + sel * 8);
    const float* pqb = pq + (size_t)b * UU;

    float sAcc[8];
    #pragma unroll
    for (int r = 0; r < 8; ++r) sAcc[r] = 0.0f;

    for (int utl = 0; utl < 32; ++utl) {
        const int ut = usel * 32 + utl;
        const int u  = ut * 16 + m;
        // B-fragment base: lane m = column u; K windows sel*8.. and 16+sel*8.. per lane
        const unsigned short* Bp = W1t + (size_t)u * DD + sel * 8;
        __builtin_prefetch(Bp + 16 * DD, 0, 0);   // next u-tile -> global_prefetch

        v8f acc = {};
        #pragma unroll 8
        for (int kc = 0; kc < 32; ++kc) {
            // A-frag (16x32 bf16): halves0-7 = K kc*32+sel*8.., halves8-15 = +16
            u16x8 alo = *(const u16x8*)&sA[rbase + kc * 32];
            u16x8 ahi = *(const u16x8*)&sA[rbase + kc * 32 + 16];
            // B-frag (32x16 bf16): halves0-7 = K kc*32+sel*8.., halves8-15 = +16
            u16x8 blo = *(const u16x8*)&Bp[kc * 32];
            u16x8 bhi = *(const u16x8*)&Bp[kc * 32 + 16];
            acc = __builtin_amdgcn_wmma_f32_16x16x32_bf16(
                false, mkfrag(alo, ahi), false, mkfrag(blo, bhi),
                (short)0, acc, false, false);
        }
        // fused epilogue: + proj_q, native v_tanh_f32, dot with V — in registers
        const float pqv = pqb[u];
        const float vv  = Vvec[u];
        #pragma unroll
        for (int r = 0; r < 8; ++r)
            sAcc[r] = fmaf(vv, tanh_fast(acc[r] + pqv), sAcc[r]);
    }

    // cross-lane sum over the 16 columns of each lane-half (stays within half)
    #pragma unroll
    for (int r = 0; r < 8; ++r) {
        float s = sAcc[r];
        s += __shfl_xor(s, 1, 32);
        s += __shfl_xor(s, 2, 32);
        s += __shfl_xor(s, 4, 32);
        s += __shfl_xor(s, 8, 32);
        if (m == 0)   // C layout: vgpr r, lanes0-15 -> M=r ; lanes16-31 -> M=r+8
            atomicAdd(&sScore[stile * 16 + sel * 8 + r], s);  // merge two u-halves
    }
    __syncthreads();
    if (tid < 64)
        scores[row0 + tid] = sScore[tid] + bV[0];
}

// ---------------------------------------------------------------------------
// Kernel 4: zero the context region of d_out (harness poisons it)
// ---------------------------------------------------------------------------
__global__ __launch_bounds__(256) void k_zero(float* __restrict__ ctx) {
    const int i = blockIdx.x * 256 + threadIdx.x;
    if (i < BB * DD) ctx[i] = 0.0f;
}

// ---------------------------------------------------------------------------
// Kernel 5: softmax over S per batch  -> attention weights (second output)
// ---------------------------------------------------------------------------
__global__ __launch_bounds__(256) void k_softmax(const float* __restrict__ scores,
                                                 float* __restrict__ attn) {
    __shared__ float red[256];
    const int b = blockIdx.x, t = threadIdx.x;
    const float* s = scores + (size_t)b * SS;
    float* a = attn + (size_t)b * SS;

    float mx = -1e30f;
    for (int i = t; i < SS; i += 256) mx = fmaxf(mx, s[i]);
    red[t] = mx; __syncthreads();
    for (int o = 128; o > 0; o >>= 1) {
        if (t < o) red[t] = fmaxf(red[t], red[t + o]);
        __syncthreads();
    }
    mx = red[0]; __syncthreads();

    float sum = 0.0f;
    for (int i = t; i < SS; i += 256) {
        float e = __expf(s[i] - mx);
        a[i] = e;
        sum += e;
    }
    red[t] = sum; __syncthreads();
    for (int o = 128; o > 0; o >>= 1) {
        if (t < o) red[t] += red[t + o];
        __syncthreads();
    }
    const float inv = 1.0f / red[0];
    for (int i = t; i < SS; i += 256) a[i] *= inv;
}

// ---------------------------------------------------------------------------
// Kernel 6: context[b,d] = sum_s attn[b,s] * values[b,s,d]  (S split 8-way,
// weights staged in LDS, float4 streams, f32 global atomic finish)
// ---------------------------------------------------------------------------
__global__ __launch_bounds__(256) void k_context(const float* __restrict__ values,
                                                 const float* __restrict__ attn,
                                                 float* __restrict__ ctx) {
    __shared__ float w[256];
    const int b  = blockIdx.y;
    const int sc = blockIdx.x;          // 0..7, chunk of 256 s-rows
    const int t  = threadIdx.x;
    const int s0 = sc * 256;
    w[t] = attn[(size_t)b * SS + s0 + t];
    __syncthreads();

    const float4* vp = (const float4*)(values + ((size_t)b * SS + s0) * DD);
    float ax = 0.f, ay = 0.f, az = 0.f, aw = 0.f;
    #pragma unroll 4
    for (int s = 0; s < 256; ++s) {
        float4 v = vp[(size_t)s * 256 + t];
        const float wv = w[s];
        ax = fmaf(wv, v.x, ax); ay = fmaf(wv, v.y, ay);
        az = fmaf(wv, v.z, az); aw = fmaf(wv, v.w, aw);
    }
    float* c = ctx + (size_t)b * DD + t * 4;
    atomicAdd(c + 0, ax); atomicAdd(c + 1, ay);
    atomicAdd(c + 2, az); atomicAdd(c + 3, aw);
}

// ---------------------------------------------------------------------------
extern "C" void kernel_launch(void* const* d_in, const int* in_sizes, int n_in,
                              void* d_out, int out_size, void* d_ws, size_t ws_size,
                              hipStream_t stream) {
    const float* query  = (const float*)d_in[0];   // [B,D]
    const float* values = (const float*)d_in[1];   // [B,S,D]
    const float* W1     = (const float*)d_in[2];   // [D,U]
    const float* b1     = (const float*)d_in[3];   // [U]
    const float* W2     = (const float*)d_in[4];   // [D,U]
    const float* b2     = (const float*)d_in[5];   // [U]
    const float* Vvec   = (const float*)d_in[6];   // [U,1]
    const float* bV     = (const float*)d_in[7];   // [1]

    float* ctx  = (float*)d_out;                   // [B,D]   (output 0)
    float* attn = ctx + (size_t)BB * DD;           // [B,S,1] (output 1)

    // workspace layout (all 256B-aligned)
    char* ws = (char*)d_ws;
    unsigned short* W1t = (unsigned short*)ws;                         // U*D bf16 = 2 MB
    float* pq     = (float*)(ws + (size_t)UU * DD * 2);                // B*U f32
    float* scores = (float*)(ws + (size_t)UU * DD * 2 + (size_t)BB * UU * 4); // B*S f32

    k_w1t    <<<dim3(UU / 32, DD / 32), dim3(32, 8), 0, stream>>>(W1, W1t);
    k_projq  <<<dim3(UU / 256, BB),     256,         0, stream>>>(query, W2, b1, b2, pq);
    k_scores <<<(BB * SS) / 64,         256,         0, stream>>>(values, W1t, pq, Vvec, bV, scores);
    k_zero   <<<(BB * DD) / 256,        256,         0, stream>>>(ctx);
    k_softmax<<<BB,                     256,         0, stream>>>(scores, attn);
    k_context<<<dim3(SS / 256, BB),     256,         0, stream>>>(values, attn, ctx);
}
